// HyperGraphAttentionLayerSparse_3427383902830
// MI455X (gfx1250) — matrix operand order repaired and checked by reading, not compiled, not run on hardware
//
#include <hip/hip_runtime.h>

typedef __attribute__((ext_vector_type(16))) __bf16 bf16x16;
typedef __attribute__((ext_vector_type(8)))  float  f32x8;

#define NB    8
#define NN1   1024
#define NN2   2048
#define FF    128
#define NEGV  (-9.0e15f)
#define LALPHA 0.2f

union frag16 { bf16x16 v; uint4 q[2]; };
union pk8    { uint4 q; __bf16 b[8]; };

__device__ __forceinline__ float lrelu(float v) { return v > 0.f ? v : LALPHA * v; }
__device__ __forceinline__ float wredmax(float v) {
#pragma unroll
    for (int o = 16; o > 0; o >>= 1) v = fmaxf(v, __shfl_xor(v, o, 32));
    return v;
}
__device__ __forceinline__ float wredsum(float v) {
#pragma unroll
    for (int o = 16; o > 0; o >>= 1) v += __shfl_xor(v, o, 32);
    return v;
}

// ---------------------------------------------------------------- k0: fold W2/W3 into vectors, W -> bf16 transposed
__global__ __launch_bounds__(256) void k0_prep(
    const float* __restrict__ W2, const float* __restrict__ W3,
    const float* __restrict__ wc, const float* __restrict__ a,
    const float* __restrict__ a2, const float* __restrict__ W,
    float* __restrict__ v1, float* __restrict__ v2, float* __restrict__ v3,
    float* __restrict__ cq, __bf16* __restrict__ W_bfT)
{
    int t = threadIdx.x;
    if (t < FF) {
        float s1 = 0.f, s2 = 0.f, s3 = 0.f;
        for (int f = 0; f < FF; ++f) {
            s1 += W2[t * FF + f] * a[FF + f];   // W2 @ a_x
            s2 += W2[t * FF + f] * a2[f];       // W2 @ a2_x
            s3 += W3[t * FF + f] * a2[FF + f];  // W3 @ a2_y
        }
        v1[t] = s1; v2[t] = s2; v3[t] = s3;
    }
    if (t == 0) {
        float s = 0.f;
        for (int f = 0; f < FF; ++f) s += wc[f] * a[f];
        *cq = s;
    }
    for (int i = t; i < FF * FF; i += 256) {
        int k = i >> 7, f = i & 127;
        W_bfT[f * FF + k] = (__bf16)W[i];       // transposed: [f][k]
    }
}

// ---------------------------------------------------------------- k1: x -> bf16 copy, s1 = lrelu(cq + x.v1), p = x.v2
__global__ __launch_bounds__(256) void k1_rowprep(
    const float* __restrict__ x, const float* __restrict__ v1,
    const float* __restrict__ v2, const float* __restrict__ cq,
    __bf16* __restrict__ x_bf, float* __restrict__ s1, float* __restrict__ p)
{
    int lane = threadIdx.x & 31, wv = threadIdx.x >> 5;
    int row = blockIdx.x * 8 + wv;  // [0, 16384)
    const float4 xv  = ((const float4*)(x + (size_t)row * FF))[lane];
    const float4 v1v = ((const float4*)v1)[lane];
    const float4 v2v = ((const float4*)v2)[lane];
    float d1 = xv.x * v1v.x + xv.y * v1v.y + xv.z * v1v.z + xv.w * v1v.w;
    float d2 = xv.x * v2v.x + xv.y * v2v.y + xv.z * v2v.z + xv.w * v2v.w;
    d1 = wredsum(d1);
    d2 = wredsum(d2);
    pk8 ob;
    ob.b[0] = (__bf16)xv.x; ob.b[1] = (__bf16)xv.y;
    ob.b[2] = (__bf16)xv.z; ob.b[3] = (__bf16)xv.w;
    // pack two lanes' worth? keep 8B store per lane: write low half only
    __bf16* xb = x_bf + (size_t)row * FF + lane * 4;
    *(uint2*)xb = *(uint2*)&ob.q;
    if (lane == 0) {
        s1[row] = lrelu(*cq + d1);
        p[row]  = d2;
    }
}

// ---------------------------------------------------------------- k2: xt = x @ W -> transposed bf16 (F, B*N2)
__global__ __launch_bounds__(256) void k2_xt_gemm(
    const __bf16* __restrict__ x_bf, const __bf16* __restrict__ W_bfT,
    __bf16* __restrict__ xt_bfT)
{
    __shared__ __bf16 WsT[128][136];   // [f][k]
    __shared__ __bf16 xs[16][136];     // [m][k]
    int t = threadIdx.x, lane = t & 31, wv = t >> 5;
    int half = lane >> 4, n = lane & 15;
    int row0 = blockIdx.x * 16;

    for (int i = t; i < 2048; i += 256) {
        int r = i >> 4, c = (i & 15) << 3;
        *(uint4*)&WsT[r][c] = *(const uint4*)&W_bfT[r * FF + c];
    }
    {
        int r = t >> 4, c = (t & 15) << 3;
        *(uint4*)&xs[r][c] = *(const uint4*)&x_bf[(size_t)(row0 + r) * FF + c];
    }
    __syncthreads();

    f32x8 acc = {0.f, 0.f, 0.f, 0.f, 0.f, 0.f, 0.f, 0.f};
#pragma unroll
    for (int c = 0; c < 4; ++c) {
        frag16 fa, fb;
        fa.q[0] = *(const uint4*)&xs[n][32 * c + 8 * half];
        fa.q[1] = *(const uint4*)&xs[n][32 * c + 8 * half + 16];
        fb.q[0] = *(const uint4*)&WsT[16 * wv + n][32 * c + 8 * half];
        fb.q[1] = *(const uint4*)&WsT[16 * wv + n][32 * c + 8 * half + 16];
        acc = __builtin_amdgcn_wmma_f32_16x16x32_bf16(false, fa.v, false, fb.v,
                                                      (short)0, acc, false, false);
    }
    pk8 ob;
#pragma unroll
    for (int g = 0; g < 8; ++g) ob.b[g] = (__bf16)acc[g];
    // transposed store: 8 consecutive rows at fixed f -> one b128
    *(uint4*)&xt_bfT[(size_t)(16 * wv + n) * (NB * NN2) + row0 + 8 * half] = ob.q;
}

// ---------------------------------------------------------------- k3: edge softmax + edge = attn @ xt (transposed out), q = edge.v3
__global__ __launch_bounds__(256) void k3_edge(
    const float* __restrict__ adj, const float* __restrict__ s1,
    const __bf16* __restrict__ xt_bfT, const float* __restrict__ v3,
    __bf16* __restrict__ edge_bfT, float* __restrict__ q)
{
    __shared__ float         s1s[NN2];
    __shared__ unsigned char maskb[16][512];    // nibble per 4 j
    __shared__ __bf16        wexp[16][1032];    // [i_local][j_local in half]
    __shared__ __bf16        xtsT[128][40];     // [f][K]
    __shared__ float         mxs[16], invs[16], v3s[FF];
    __shared__ float         qpart[8][16];

    int t = threadIdx.x, lane = t & 31, wv = t >> 5;
    int half = lane >> 4, n = lane & 15;
    int b = blockIdx.y, i0 = blockIdx.x * 16;

    for (int i = t; i < NN2; i += 256) s1s[i] = s1[(size_t)b * NN2 + i];
    if (t < FF) v3s[t] = v3[t];
    __syncthreads();

    const float* adjb = adj + ((size_t)b * NN1 + i0) * NN2;

    // pass 1: mask nibbles + row max (2 rows per wave, float4 adj loads)
    for (int rr = 0; rr < 2; ++rr) {
        int row = wv * 2 + rr;
        const float4* ar = (const float4*)(adjb + (size_t)row * NN2);
        float mx = NEGV;
        for (int c2 = 0; c2 < 16; ++c2) {
            int j = c2 * 128 + lane * 4;
            float4 av = ar[c2 * 32 + lane];
            unsigned nib = 0u;
            if (av.x > 0.f) { nib |= 1u; mx = fmaxf(mx, s1s[j + 0]); }
            if (av.y > 0.f) { nib |= 2u; mx = fmaxf(mx, s1s[j + 1]); }
            if (av.z > 0.f) { nib |= 4u; mx = fmaxf(mx, s1s[j + 2]); }
            if (av.w > 0.f) { nib |= 8u; mx = fmaxf(mx, s1s[j + 3]); }
            maskb[row][c2 * 32 + lane] = (unsigned char)nib;
        }
        mx = wredmax(mx);
        if (lane == 0) mxs[row] = mx;
    }
    __syncthreads();

    f32x8 acc = {0.f, 0.f, 0.f, 0.f, 0.f, 0.f, 0.f, 0.f};
    float sums0 = 0.f, sums1 = 0.f;

    for (int h = 0; h < 2; ++h) {
        __syncthreads();   // wexp reuse guard
        // pass 2(h): bf16 exp weights + partial sums (2 rows per wave)
        for (int rr = 0; rr < 2; ++rr) {
            int row = wv * 2 + rr;
            float mx = mxs[row];
            float s = 0.f;
            for (int c = 0; c < 32; ++c) {
                int jl = c * 32 + lane;
                int j  = h * 1024 + jl;
                unsigned nib = maskb[row][j >> 2];
                float e = ((nib >> (j & 3)) & 1u) ? __expf(s1s[j] - mx) : 0.f;
                wexp[row][jl] = (__bf16)e;
                s += e;
            }
            s = wredsum(s);
            if (rr == 0) sums0 += s; else sums1 += s;
        }
        __syncthreads();
        // WMMA over this K-half: 32 chunks of 32
        for (int c = 0; c < 32; ++c) {
            __syncthreads();
#pragma unroll
            for (int it = 0; it < 2; ++it) {    // stage 128(f) x 32(K) bf16
                int idx = t + 256 * it;
                int f = idx >> 2, pp = idx & 3;
                *(uint4*)&xtsT[f][pp * 8] =
                    *(const uint4*)&xt_bfT[(size_t)f * (NB * NN2) + (size_t)b * NN2 +
                                           h * 1024 + c * 32 + pp * 8];
            }
            __syncthreads();
            frag16 fa, fb;
            fa.q[0] = *(const uint4*)&wexp[n][c * 32 + 8 * half];
            fa.q[1] = *(const uint4*)&wexp[n][c * 32 + 8 * half + 16];
            fb.q[0] = *(const uint4*)&xtsT[16 * wv + n][8 * half];
            fb.q[1] = *(const uint4*)&xtsT[16 * wv + n][8 * half + 16];
            acc = __builtin_amdgcn_wmma_f32_16x16x32_bf16(false, fa.v, false, fb.v,
                                                          (short)0, acc, false, false);
        }
    }

    if (lane == 0) {
        invs[wv * 2 + 0] = (sums0 > 0.f) ? (1.f / sums0) : 0.f;
        invs[wv * 2 + 1] = (sums1 > 0.f) ? (1.f / sums1) : 0.f;
    }
    __syncthreads();

    // epilogue: normalize, transposed bf16 edge store, deterministic q = edge.v3
    int f = 16 * wv + n;
    float v3f = v3s[f];
    pk8 ob;
#pragma unroll
    for (int g = 0; g < 8; ++g) {
        int r = g + 8 * half;
        float val = acc[g] * invs[r];
        ob.b[g] = (__bf16)val;
        float qp = val * v3f;
#pragma unroll
        for (int o = 1; o < 16; o <<= 1) qp += __shfl_xor(qp, o, 32);
        if (n == 0) qpart[wv][r] = qp;
    }
    *(uint4*)&edge_bfT[(size_t)f * (NB * NN1) + (size_t)b * NN1 + i0 + 8 * half] = ob.q;
    __syncthreads();
    if (t < 16) {
        float s = 0.f;
        for (int w = 0; w < 8; ++w) s += qpart[w][t];
        q[(size_t)b * NN1 + i0 + t] = s;
    }
}

// ---------------------------------------------------------------- k4: node softmax (transposed mask) + node = attn @ edge, ELU
__global__ __launch_bounds__(256) void k4_node(
    const float* __restrict__ adj, const float* __restrict__ p,
    const float* __restrict__ q, const __bf16* __restrict__ edge_bfT,
    float* __restrict__ out)
{
    __shared__ float  qs[NN1];
    __shared__ __bf16 scb[16][1032];   // [j_local][i] scores -> exp weights
    __shared__ __bf16 esT[128][40];    // [f][K]
    __shared__ float  red[64][16];
    __shared__ float  mxs[16], invs[16], pv[16];
    __shared__ float  ot[16][132];     // output transpose staging

    int t = threadIdx.x, lane = t & 31, wv = t >> 5;
    int half = lane >> 4, n = lane & 15;
    int b = blockIdx.y, j0 = blockIdx.x * 16;

    for (int i = t; i < NN1; i += 256) qs[i] = q[(size_t)b * NN1 + i];
    if (t < 16) pv[t] = p[(size_t)b * NN2 + j0 + t];
    __syncthreads();

    const float* adjb = adj + (size_t)b * NN1 * NN2;
    int tq = t >> 2, jq = (t & 3) * 4;   // thread covers 4 j columns, 16 i's

    // pass 1: scores -> scb (bf16), column max over i
    float mx0 = NEGV, mx1 = NEGV, mx2 = NEGV, mx3 = NEGV;
    for (int k = 0; k < 16; ++k) {
        int i = k * 64 + tq;
        float4 av = *(const float4*)&adjb[(size_t)i * NN2 + j0 + jq];
        float qv = qs[i];
        float s0 = (av.x > 0.f) ? lrelu(pv[jq + 0] + qv) : NEGV;
        float s1v = (av.y > 0.f) ? lrelu(pv[jq + 1] + qv) : NEGV;
        float s2 = (av.z > 0.f) ? lrelu(pv[jq + 2] + qv) : NEGV;
        float s3 = (av.w > 0.f) ? lrelu(pv[jq + 3] + qv) : NEGV;
        scb[jq + 0][i] = (__bf16)s0; scb[jq + 1][i] = (__bf16)s1v;
        scb[jq + 2][i] = (__bf16)s2; scb[jq + 3][i] = (__bf16)s3;
        mx0 = fmaxf(mx0, s0); mx1 = fmaxf(mx1, s1v);
        mx2 = fmaxf(mx2, s2); mx3 = fmaxf(mx3, s3);
    }
    red[tq][jq + 0] = mx0; red[tq][jq + 1] = mx1;
    red[tq][jq + 2] = mx2; red[tq][jq + 3] = mx3;
    __syncthreads();
    if (t < 16) {
        float m = red[0][t];
        for (int r = 1; r < 64; ++r) m = fmaxf(m, red[r][t]);
        mxs[t] = m;
    }
    __syncthreads();

    // pass 2: exp in place + column sums
    float su0 = 0.f, su1 = 0.f, su2 = 0.f, su3 = 0.f;
    float m0 = mxs[jq + 0], m1 = mxs[jq + 1], m2 = mxs[jq + 2], m3 = mxs[jq + 3];
    for (int k = 0; k < 16; ++k) {
        int i = k * 64 + tq;
        float e0 = __expf((float)scb[jq + 0][i] - m0);
        float e1 = __expf((float)scb[jq + 1][i] - m1);
        float e2 = __expf((float)scb[jq + 2][i] - m2);
        float e3 = __expf((float)scb[jq + 3][i] - m3);
        scb[jq + 0][i] = (__bf16)e0; scb[jq + 1][i] = (__bf16)e1;
        scb[jq + 2][i] = (__bf16)e2; scb[jq + 3][i] = (__bf16)e3;
        su0 += e0; su1 += e1; su2 += e2; su3 += e3;
    }
    red[tq][jq + 0] = su0; red[tq][jq + 1] = su1;
    red[tq][jq + 2] = su2; red[tq][jq + 3] = su3;
    __syncthreads();
    if (t < 16) {
        float s = 0.f;
        for (int r = 0; r < 64; ++r) s += red[r][t];
        invs[t] = (s > 0.f) ? (1.f / s) : 0.f;
    }
    __syncthreads();

    // WMMA: node_unnorm[j, f] += w[j, i] * edge[i, f]
    f32x8 acc = {0.f, 0.f, 0.f, 0.f, 0.f, 0.f, 0.f, 0.f};
    for (int c = 0; c < 32; ++c) {
        __syncthreads();
#pragma unroll
        for (int it = 0; it < 2; ++it) {    // stage 128(f) x 32(K=i)
            int idx = t + 256 * it;
            int f = idx >> 2, pp = idx & 3;
            *(uint4*)&esT[f][pp * 8] =
                *(const uint4*)&edge_bfT[(size_t)f * (NB * NN1) + (size_t)b * NN1 +
                                         c * 32 + pp * 8];
        }
        __syncthreads();
        frag16 fa, fb;
        fa.q[0] = *(const uint4*)&scb[n][c * 32 + 8 * half];
        fa.q[1] = *(const uint4*)&scb[n][c * 32 + 8 * half + 16];
        fb.q[0] = *(const uint4*)&esT[16 * wv + n][8 * half];
        fb.q[1] = *(const uint4*)&esT[16 * wv + n][8 * half + 16];
        acc = __builtin_amdgcn_wmma_f32_16x16x32_bf16(false, fa.v, false, fb.v,
                                                      (short)0, acc, false, false);
    }

    // epilogue: normalize + ELU -> LDS transpose -> coalesced fp32 stores
#pragma unroll
    for (int g = 0; g < 8; ++g) {
        int r = g + 8 * half;
        float val = acc[g] * invs[r];
        ot[r][16 * wv + n] = (val > 0.f) ? val : (__expf(val) - 1.f);
    }
    __syncthreads();
    {
        int r = t >> 4, cc = (t & 15) * 8;
        float4* dst = (float4*)&out[((size_t)b * NN2 + j0 + r) * FF + cc];
        dst[0] = *(float4*)&ot[r][cc];
        dst[1] = *(float4*)&ot[r][cc + 4];
    }
}

// ---------------------------------------------------------------- launch
extern "C" void kernel_launch(void* const* d_in, const int* in_sizes, int n_in,
                              void* d_out, int out_size, void* d_ws, size_t ws_size,
                              hipStream_t stream)
{
    const float* x   = (const float*)d_in[0];
    const float* adj = (const float*)d_in[1];
    const float* W   = (const float*)d_in[2];
    const float* W2  = (const float*)d_in[3];
    const float* W3  = (const float*)d_in[4];
    const float* wc  = (const float*)d_in[5];
    const float* a   = (const float*)d_in[6];
    const float* a2  = (const float*)d_in[7];
    float* out = (float*)d_out;
    char* ws = (char*)d_ws;

    const size_t OFF_V1   = 0;
    const size_t OFF_V2   = 512;
    const size_t OFF_V3   = 1024;
    const size_t OFF_CQ   = 1536;
    const size_t OFF_S1   = 2048;                       // 65536
    const size_t OFF_P    = OFF_S1 + 65536;             // 65536
    const size_t OFF_Q    = OFF_P + 65536;              // 32768
    const size_t OFF_WBF  = OFF_Q + 32768;              // 32768
    const size_t OFF_XBF  = OFF_WBF + 32768;            // 4194304
    const size_t OFF_XTBF = OFF_XBF + 4194304;          // 4194304
    const size_t OFF_EBF  = OFF_XTBF + 4194304;         // 2097152
    const size_t NEED     = OFF_EBF + 2097152;
    if (ws_size < NEED) return;

    float*  v1      = (float*)(ws + OFF_V1);
    float*  v2      = (float*)(ws + OFF_V2);
    float*  v3      = (float*)(ws + OFF_V3);
    float*  cq      = (float*)(ws + OFF_CQ);
    float*  s1      = (float*)(ws + OFF_S1);
    float*  pvec    = (float*)(ws + OFF_P);
    float*  qvec    = (float*)(ws + OFF_Q);
    __bf16* W_bfT   = (__bf16*)(ws + OFF_WBF);
    __bf16* x_bf    = (__bf16*)(ws + OFF_XBF);
    __bf16* xt_bfT  = (__bf16*)(ws + OFF_XTBF);
    __bf16* edge_bfT= (__bf16*)(ws + OFF_EBF);

    k0_prep<<<1, 256, 0, stream>>>(W2, W3, wc, a, a2, W, v1, v2, v3, cq, W_bfT);
    k1_rowprep<<<2048, 256, 0, stream>>>(x, v1, v2, cq, x_bf, s1, pvec);
    k2_xt_gemm<<<1024, 256, 0, stream>>>(x_bf, W_bfT, xt_bfT);
    dim3 g3(NN1 / 16, NB);
    k3_edge<<<g3, 256, 0, stream>>>(adj, s1, xt_bfT, v3, edge_bfT, qvec);
    dim3 g4(NN2 / 16, NB);
    k4_node<<<g4, 256, 0, stream>>>(adj, pvec, qvec, edge_bfT, out);
}
